// DiffusedFarthestAttention_86809878987153
// MI455X (gfx1250) — matrix operand order repaired and checked by reading, not compiled, not run on hardware
//
#include <hip/hip_runtime.h>
#include <hip/hip_bf16.h>
#include <math.h>

// ---------------- types ----------------
typedef __bf16 bf16_t;
typedef __attribute__((ext_vector_type(16))) __bf16 v16bf;
typedef __attribute__((ext_vector_type(8)))  __bf16 v8bf;
typedef __attribute__((ext_vector_type(8)))  float  v8f;
typedef __attribute__((ext_vector_type(4))) unsigned int u32x4;
typedef __attribute__((ext_vector_type(8))) int i32x8;
typedef __attribute__((ext_vector_type(4))) int i32x4;

// ---------------- problem sizes ----------------
#define WB 4
#define WN 32768
#define WK 128
#define WM 1024
#define WC 256
#define WH 8
#define WD 32
#define NCH1 64
#define CHN1 512
#define NCH2 8

__device__ __forceinline__ bf16_t f2bf(float x) { return (bf16_t)x; }

// ---------------- WMMA helpers (CDNA5 gfx1250, wave32) ----------------
// A-frag (16x32 MxK, bf16): lane l holds row (l&15); lanes<16: K=0..7,16..23 ; lanes>=16: K=8..15,24..31
__device__ __forceinline__ v16bf load_a(const bf16_t* row0, int stride) {
  int lane = threadIdx.x & 31;
  const bf16_t* p = row0 + (lane & 15) * stride + ((lane >> 4) << 3);
  v8bf lo = *(const v8bf*)(p);
  v8bf hi = *(const v8bf*)(p + 16);
  v16bf a;
#pragma unroll
  for (int e = 0; e < 8; ++e) { a[e] = lo[e]; a[8 + e] = hi[e]; }
  return a;
}
// B-frag (32x16 KxN, bf16) from B^T row-major storage ([col][k]):
// lane l holds col (l&15); lanes<16: K=0..7,16..23 ; lanes>=16: K=8..15,24..31
__device__ __forceinline__ v16bf load_b(const bf16_t* col0, int stride) {
  int lane = threadIdx.x & 31;
  const bf16_t* p = col0 + (lane & 15) * stride + ((lane >> 4) << 3);
  v8bf lo = *(const v8bf*)(p);
  v8bf hi = *(const v8bf*)(p + 16);
  v16bf b;
#pragma unroll
  for (int e = 0; e < 8; ++e) { b[e] = lo[e]; b[8 + e] = hi[e]; }
  return b;
}
__device__ __forceinline__ v8f wmma_bf16(v16bf a, v16bf b, v8f c) {
  return __builtin_amdgcn_wmma_f32_16x16x32_bf16(false, a, false, b, (short)0, c, false, false);
}
// C/D layout: VGPR j: lanes<16 -> row j, lanes>=16 -> row j+8; col = lane&15
__device__ __forceinline__ void store_cd_f32(float* base, int stride, v8f c) {
  int lane = threadIdx.x & 31;
  float* p = base + ((lane >> 4) << 3) * stride + (lane & 15);
#pragma unroll
  for (int j = 0; j < 8; ++j) p[j * stride] = c[j];
}

// ---------------- Tensor Data Mover: 2D bf16 tile -> LDS with row padding ----------------
// rows x row_elems bf16 tile, source line stride stride_elems, LDS rows padded by
// (pad_a+1)*4 bytes after every 2^(pad_i+1) DWORDs (== one row of data).
__device__ __forceinline__ void tdm_load_2d_bf16(
    void* ldsdst, const void* gsrc, unsigned rows, unsigned row_elems,
    unsigned stride_elems, unsigned pad_i, unsigned pad_a) {
  unsigned long long ga = (unsigned long long)(uintptr_t)gsrc;
  unsigned lds = (unsigned)(uintptr_t)ldsdst;
  u32x4 g0;
  g0[0] = 1u;                                            // count=1 (user descriptor)
  g0[1] = lds;                                           // lds_addr
  g0[2] = (unsigned)(ga & 0xffffffffu);                  // global_addr[31:0]
  g0[3] = (unsigned)((ga >> 32) & 0x1ffffffu) | (2u << 30); // global_addr[56:32], type=2
  i32x8 g1;
  g1[0] = (int)((1u << 16) | (1u << 20) | (pad_i << 22) | (pad_a << 25)); // 2B elems, pad_enable
  g1[1] = (int)((row_elems & 0xffffu) << 16);            // tensor_dim0[15:0]
  g1[2] = (int)(((row_elems >> 16) & 0xffffu) | ((rows & 0xffffu) << 16)); // dim0 hi | tensor_dim1 lo
  g1[3] = (int)(((rows >> 16) & 0xffffu) | ((row_elems & 0xffffu) << 16)); // dim1 hi | tile_dim0
  g1[4] = (int)(rows & 0xffffu);                         // tile_dim1 (tile_dim2=0)
  g1[5] = (int)stride_elems;                             // tensor_dim0_stride[31:0]
  g1[6] = 0;
  g1[7] = 0;
  i32x4 z4 = {0, 0, 0, 0};
  i32x8 z8 = {0, 0, 0, 0, 0, 0, 0, 0};
  __builtin_amdgcn_tensor_load_to_lds(g0, g1, z4, z4, z8, 0);
}
__device__ __forceinline__ void tdm_wait() { __builtin_amdgcn_s_wait_tensorcnt(0); }

// ---------------- K1: to_basis1 partial:  part[b][ch][k][c] = sum_n evecs[n][k]*mass[n]*x[n][c] ----------------
__global__ __launch_bounds__(256) void k_tobasis1(
    const float* __restrict__ x, const float* __restrict__ mass,
    const float* __restrict__ evecs, float* __restrict__ part) {
  __shared__ bf16_t aT[WK * 40];   // [k][n] , n-tile of 32, stride 40
  __shared__ bf16_t bT[WC * 40];   // [c][n]
  __shared__ float  msS[CHN1];
  int b = blockIdx.y, ch = blockIdx.x;
  int n0 = ch * CHN1;
  int t = threadIdx.x, w = t >> 5;
  for (int i = t; i < CHN1; i += 256) msS[i] = mass[(long)b * WN + n0 + i];
  v8f zero = {};
  v8f acc[16];
#pragma unroll
  for (int i = 0; i < 16; ++i) acc[i] = zero;
  __syncthreads();
  for (int ns = 0; ns < CHN1; ns += 32) {
    for (int i = 0; i < 16; ++i) {           // 32x128 evecs -> aT transposed
      int e = t + 256 * i; int nn = e >> 7, k = e & 127;
      const float* gp = &evecs[((long)b * WN + n0 + ns + nn) * WK + k];
      if (ns + 64 < CHN1) __builtin_prefetch(gp + 64 * WK, 0, 1);
      aT[k * 40 + nn] = f2bf(*gp);
    }
    for (int i = 0; i < 32; ++i) {           // 32x256 (mass*x) -> bT transposed
      int e = t + 256 * i; int nn = e >> 8, c = e & 255;
      const float* gp = &x[((long)b * WN + n0 + ns + nn) * WC + c];
      if (ns + 64 < CHN1) __builtin_prefetch(gp + 64 * WC, 0, 1);
      bT[c * 40 + nn] = f2bf(*gp * msS[ns + nn]);
    }
    __syncthreads();
    v16bf af = load_a(&aT[(w * 16) * 40], 40);
    v16bf bcur = load_b(&bT[0], 40);
#pragma unroll
    for (int ct = 0; ct < 16; ++ct) {
      v16bf bnxt = bcur;
      if (ct < 15) bnxt = load_b(&bT[((ct + 1) * 16) * 40], 40);
      acc[ct] = wmma_bf16(af, bcur, acc[ct]);
      bcur = bnxt;
    }
    __syncthreads();
  }
  float* pout = part + ((long)b * NCH1 + ch) * WK * WC;
#pragma unroll
  for (int ct = 0; ct < 16; ++ct)
    store_cd_f32(pout + (w * 16) * WC + ct * 16, WC, acc[ct]);
}

// ---------------- K2: reduce partials, apply exp(-evals*t), emit bf16 spec TRANSPOSED [b][c][k] ----------------
__global__ __launch_bounds__(256) void k_reduce(
    const float* __restrict__ part, const float* __restrict__ evals,
    const float* __restrict__ tvec, bf16_t* __restrict__ specT, int nch) {
  long idx = (long)blockIdx.x * 256 + threadIdx.x;   // over B*K*C
  int c = (int)(idx & 255); long bk = idx >> 8; int k = (int)(bk & 127); int b = (int)(bk >> 7);
  float s = 0.f;
  const float* p = part + (long)b * nch * WK * WC + (long)k * WC + c;
  for (int i = 0; i < nch; ++i) s += p[(long)i * WK * WC];
  float tt = fmaxf(tvec[c], 1e-8f);
  float coef = __expf(-evals[b * WK + k] * tt);
  specT[((long)b * WC + c) * WK + k] = f2bf(s * coef);
}

// ---------------- K3: from_basis at gathered far rows (B-operand staged by TDM) ----------------
__global__ __launch_bounds__(256) void k_frombasis_far(
    const float* __restrict__ evecs, const int* __restrict__ far,
    const bf16_t* __restrict__ specT, float* __restrict__ xfar) {
  __shared__ bf16_t aS[128 * 136];  // [m][k]
  __shared__ bf16_t bS[256 * 136];  // [c][k]  (TDM, 256B data + 16B pad per row)
  int b = blockIdx.y, mt = blockIdx.x; int m0 = mt * 128;
  int t = threadIdx.x, w = t >> 5;
  if (t < 32) tdm_load_2d_bf16(bS, specT + (long)b * WC * WK, WC, WK, WK, 5, 3);
  for (int i = 0; i < 64; ++i) {
    int e = t + 256 * i; int mm = e >> 7, k = e & 127;
    int row = far[b * WM + m0 + mm];
    aS[mm * 136 + k] = f2bf(evecs[((long)b * WN + row) * WK + k]);
  }
  if (t < 32) tdm_wait();
  __syncthreads();
  v8f zero = {}; v8f acc[16];
#pragma unroll
  for (int i = 0; i < 16; ++i) acc[i] = zero;
  for (int k0 = 0; k0 < 128; k0 += 32) {
    v16bf af = load_a(&aS[(w * 16) * 136 + k0], 136);
    v16bf bcur = load_b(&bS[k0], 136);
#pragma unroll
    for (int ct = 0; ct < 16; ++ct) {
      v16bf bnxt = bcur;
      if (ct < 15) bnxt = load_b(&bS[((ct + 1) * 16) * 136 + k0], 136);
      acc[ct] = wmma_bf16(af, bcur, acc[ct]);
      bcur = bnxt;
    }
  }
  float* o = xfar + ((long)b * WM + m0) * WC;
#pragma unroll
  for (int ct = 0; ct < 16; ++ct)
    store_cd_f32(o + (w * 16) * WC + ct * 16, WC, acc[ct]);
}

// ---------------- K4: GroupNorm statistics per (b,g) over 1024*8 elems ----------------
__global__ __launch_bounds__(256) void k_gnstats(
    const float* __restrict__ xfar, float* __restrict__ stats) {
  __shared__ float s1[256], s2[256];
  int b = blockIdx.x >> 5, g = blockIdx.x & 31;
  int t = threadIdx.x;
  float a = 0.f, q = 0.f;
  for (int e = t; e < WM * 8; e += 256) {
    int m = e >> 3, chn = e & 7;
    float v = xfar[((long)b * WM + m) * WC + g * 8 + chn];
    a += v; q += v * v;
  }
  s1[t] = a; s2[t] = q; __syncthreads();
  for (int st = 128; st > 0; st >>= 1) {
    if (t < st) { s1[t] += s1[t + st]; s2[t] += s2[t + st]; }
    __syncthreads();
  }
  if (t == 0) {
    float mu = s1[0] / (WM * 8.f);
    float var = s2[0] / (WM * 8.f) - mu * mu;
    stats[(b * 32 + g) * 2 + 0] = mu;
    stats[(b * 32 + g) * 2 + 1] = rsqrtf(var + 1e-6f);
  }
}

// ---------------- K5: apply GroupNorm, emit bf16 ----------------
__global__ __launch_bounds__(256) void k_gnapply(
    const float* __restrict__ xfar, const float* __restrict__ stats,
    const float* __restrict__ gw, const float* __restrict__ gb,
    bf16_t* __restrict__ xn) {
  long idx = (long)blockIdx.x * 256 + threadIdx.x;   // over B*M*C
  int c = (int)(idx & 255); int b = (int)(idx >> 18);
  int g = c >> 3;
  float mu = stats[(b * 32 + g) * 2], rs = stats[(b * 32 + g) * 2 + 1];
  xn[idx] = f2bf((xfar[idx] - mu) * rs * gw[c] + gb[c]);
}

// ---------------- K5b: pre-transpose weights to bf16 [co][ci] for TDM staging ----------------
__global__ __launch_bounds__(256) void k_wprep(
    const float* __restrict__ Wq, const float* __restrict__ Wk,
    const float* __restrict__ Wv, const float* __restrict__ Wo,
    bf16_t* __restrict__ wT) {
  int idx = blockIdx.x * 256 + threadIdx.x;  // 4*65536
  int which = idx >> 16, rem = idx & 65535;
  int co = rem >> 8, ci = rem & 255;
  const float* W = which == 0 ? Wq : (which == 1 ? Wk : (which == 2 ? Wv : Wo));
  wT[(long)which * 65536 + co * 256 + ci] = f2bf(W[ci * 256 + co]);
}

// ---------------- K6: Q/K/V projections; both operands staged by TDM ----------------
__global__ __launch_bounds__(256) void k_qkv(
    const bf16_t* __restrict__ xn, const bf16_t* __restrict__ wT,
    const float* __restrict__ bq, const float* __restrict__ bk,
    const float* __restrict__ bv,
    bf16_t* __restrict__ qb, bf16_t* __restrict__ kb, bf16_t* __restrict__ vb) {
  __shared__ bf16_t aS[128 * 264];  // [m][cin]   (512B data + 16B pad per row)
  __shared__ bf16_t bS[256 * 264];  // [cout][cin]
  int b = blockIdx.z, which = blockIdx.y, mt = blockIdx.x; int m0 = mt * 128;
  const float* bias = which == 0 ? bq : (which == 1 ? bk : bv);
  bf16_t* outp = which == 0 ? qb : (which == 1 ? kb : vb);
  int t = threadIdx.x, w = t >> 5;
  if (t < 32) {
    tdm_load_2d_bf16(aS, xn + ((long)b * WM + m0) * WC, 128, WC, WC, 6, 3);
    tdm_load_2d_bf16(bS, wT + (long)which * 65536, WC, WC, WC, 6, 3);
    tdm_wait();
  }
  __syncthreads();
  v8f zero = {}; v8f acc[16];
#pragma unroll
  for (int i = 0; i < 16; ++i) acc[i] = zero;
  for (int k0 = 0; k0 < 256; k0 += 32) {
    v16bf af = load_a(&aS[(w * 16) * 264 + k0], 264);
    v16bf bcur = load_b(&bS[k0], 264);
#pragma unroll
    for (int ct = 0; ct < 16; ++ct) {
      v16bf bnxt = bcur;
      if (ct < 15) bnxt = load_b(&bS[((ct + 1) * 16) * 264 + k0], 264);
      acc[ct] = wmma_bf16(af, bcur, acc[ct]);
      bcur = bnxt;
    }
  }
  float scale = (which == 0) ? 0.17677669529663687f : 1.0f;  // 1/sqrt(d) folded into Q
  int lane = t & 31; int col = lane & 15; int r0 = (lane >> 4) * 8;
#pragma unroll
  for (int ct = 0; ct < 16; ++ct) {
    int co = ct * 16 + col; int h = co >> 5, d = co & 31;
    float bi = bias[co];
#pragma unroll
    for (int j = 0; j < 8; ++j) {
      int m = m0 + w * 16 + r0 + j;
      outp[(((long)b * WH + h) * WM + m) * WD + d] = f2bf((acc[ct][j] + bi) * scale);
    }
  }
}

// ---------------- K7: attention, scores kept entirely in LDS (two-pass softmax) ----------------
__global__ __launch_bounds__(256) void k_attn(
    const bf16_t* __restrict__ qb, const bf16_t* __restrict__ kb,
    const bf16_t* __restrict__ vb, bf16_t* __restrict__ ao) {
  __shared__ float  sc[32 * 1032];     // scores [32 q][1024 keys]
  __shared__ bf16_t pS[32 * 1032];     // exp(S - max) in bf16
  __shared__ bf16_t vT[32 * 1032];     // V^T [d][key]
  __shared__ bf16_t qS[32 * 40];       // Q tile [q][d]
  __shared__ float  rsum[32];
  int b = blockIdx.z, h = blockIdx.y, qt = blockIdx.x; int m0 = qt * 32;
  int t = threadIdx.x, w = t >> 5, lane = t & 31;
  const bf16_t* qp = qb + ((long)b * WH + h) * WM * WD;
  const bf16_t* kp = kb + ((long)b * WH + h) * WM * WD;
  const bf16_t* vp = vb + ((long)b * WH + h) * WM * WD;
  v8f zero = {};
  for (int i = 0; i < 4; ++i) {            // stage Q tile
    int e = t + 256 * i; int mq = e >> 5, d = e & 31;
    qS[mq * 40 + d] = qp[(long)(m0 + mq) * WD + d];
  }
  for (int i = 0; i < 128; ++i) {          // stage V^T
    int e = t + 256 * i; int key = e >> 5, d = e & 31;
    vT[d * 1032 + key] = vp[(long)key * WD + d];
  }
  __syncthreads();
  // Phase B: S = Q @ K^T  (K-frag loaded straight from global, [key][d] rows)
  {
    int rt = w & 1, ktb = w >> 1;
    v16bf af = load_a(&qS[(rt * 16) * 40], 40);
    for (int kt = ktb; kt < 64; kt += 4) {
      v16bf bfr = load_b(kp + (long)(kt * 16) * WD, WD);
      v8f s = wmma_bf16(af, bfr, zero);
      store_cd_f32(&sc[(rt * 16) * 1032 + kt * 16], 1032, s);
    }
  }
  __syncthreads();
  // Phase C: row softmax (unnormalized p, keep row sums)
  for (int rr = 0; rr < 4; ++rr) {
    int r = w * 4 + rr;
    float mx = -1e30f;
    for (int i = lane; i < 1024; i += 32) mx = fmaxf(mx, sc[r * 1032 + i]);
#pragma unroll
    for (int o = 16; o > 0; o >>= 1) mx = fmaxf(mx, __shfl_xor(mx, o, 32));
    float sm = 0.f;
    for (int i = lane; i < 1024; i += 32) {
      float e = __expf(sc[r * 1032 + i] - mx);
      sm += e;
      pS[r * 1032 + i] = f2bf(e);
    }
#pragma unroll
    for (int o = 16; o > 0; o >>= 1) sm += __shfl_xor(sm, o, 32);
    if (lane == 0) rsum[r] = sm;
  }
  __syncthreads();
  // Phase D: O = P @ V  (waves 0..3: tiles (rowtile, dtile)), normalize rows on store
  if (w < 4) {
    int rt2 = w & 1, dt = w >> 1;
    v8f oacc = zero;
    v16bf pa = load_a(&pS[(rt2 * 16) * 1032], 1032);
    v16bf vf = load_b(&vT[(dt * 16) * 1032], 1032);
    for (int k0 = 0; k0 < 1024; k0 += 32) {
      v16bf pn = pa, vn = vf;
      if (k0 < 992) {
        pn = load_a(&pS[(rt2 * 16) * 1032 + k0 + 32], 1032);
        vn = load_b(&vT[(dt * 16) * 1032 + k0 + 32], 1032);
      }
      oacc = wmma_bf16(pa, vf, oacc);
      pa = pn; vf = vn;
    }
    int col = lane & 15, r0 = (lane >> 4) * 8;
#pragma unroll
    for (int j = 0; j < 8; ++j) {
      int r = rt2 * 16 + r0 + j;
      float v = oacc[j] / rsum[r];
      int m = m0 + r, c = h * 32 + dt * 16 + col;
      ao[((long)b * WM + m) * WC + c] = f2bf(v);
    }
  }
}

// ---------------- K8: output projection Wo + bias, fold mass[far[m]]; TDM staging ----------------
__global__ __launch_bounds__(256) void k_oproj(
    const bf16_t* __restrict__ ao, const bf16_t* __restrict__ woT,
    const float* __restrict__ bo, const float* __restrict__ mass,
    const int* __restrict__ far, bf16_t* __restrict__ yfar) {
  __shared__ bf16_t aS[128 * 264];
  __shared__ bf16_t bS[256 * 264];
  int b = blockIdx.y, mt = blockIdx.x; int m0 = mt * 128;
  int t = threadIdx.x, w = t >> 5;
  if (t < 32) {
    tdm_load_2d_bf16(aS, ao + ((long)b * WM + m0) * WC, 128, WC, WC, 6, 3);
    tdm_load_2d_bf16(bS, woT, WC, WC, WC, 6, 3);
    tdm_wait();
  }
  __syncthreads();
  v8f zero = {}; v8f acc[16];
#pragma unroll
  for (int i = 0; i < 16; ++i) acc[i] = zero;
  for (int k0 = 0; k0 < 256; k0 += 32) {
    v16bf af = load_a(&aS[(w * 16) * 264 + k0], 264);
    v16bf bcur = load_b(&bS[k0], 264);
#pragma unroll
    for (int ct = 0; ct < 16; ++ct) {
      v16bf bnxt = bcur;
      if (ct < 15) bnxt = load_b(&bS[((ct + 1) * 16) * 264 + k0], 264);
      acc[ct] = wmma_bf16(af, bcur, acc[ct]);
      bcur = bnxt;
    }
  }
  int lane = t & 31, col = lane & 15, r0 = (lane >> 4) * 8;
#pragma unroll
  for (int ct = 0; ct < 16; ++ct) {
    int co = ct * 16 + col; float bi = bo[co];
#pragma unroll
    for (int j = 0; j < 8; ++j) {
      int m = m0 + w * 16 + r0 + j;
      float mw = mass[(long)b * WN + far[b * WM + m]];
      yfar[((long)b * WM + m) * WC + co] = f2bf((acc[ct][j] + bi) * mw);
    }
  }
}

// ---------------- K9: to_basis2 over far rows only (z is zero elsewhere) ----------------
__global__ __launch_bounds__(256) void k_tobasis2(
    const bf16_t* __restrict__ yfar, const float* __restrict__ evecs,
    const int* __restrict__ far, float* __restrict__ part) {
  __shared__ bf16_t aT[WK * 40];   // [k][m]
  __shared__ bf16_t bT[WC * 40];   // [c][m]
  int b = blockIdx.y, ch = blockIdx.x; int m0 = ch * 128;
  int t = threadIdx.x, w = t >> 5;
  v8f zero = {}; v8f acc[16];
#pragma unroll
  for (int i = 0; i < 16; ++i) acc[i] = zero;
  for (int ms = 0; ms < 128; ms += 32) {
    __syncthreads();
    for (int i = 0; i < 16; ++i) {
      int e = t + 256 * i; int mm = e >> 7, k = e & 127;
      int row = far[b * WM + m0 + ms + mm];
      aT[k * 40 + mm] = f2bf(evecs[((long)b * WN + row) * WK + k]);
    }
    for (int i = 0; i < 32; ++i) {
      int e = t + 256 * i; int mm = e >> 8, c = e & 255;
      bT[c * 40 + mm] = yfar[((long)b * WM + m0 + ms + mm) * WC + c];
    }
    __syncthreads();
    v16bf af = load_a(&aT[(w * 16) * 40], 40);
    v16bf bcur = load_b(&bT[0], 40);
#pragma unroll
    for (int ct = 0; ct < 16; ++ct) {
      v16bf bnxt = bcur;
      if (ct < 15) bnxt = load_b(&bT[((ct + 1) * 16) * 40], 40);
      acc[ct] = wmma_bf16(af, bcur, acc[ct]);
      bcur = bnxt;
    }
  }
  float* pout = part + ((long)b * NCH2 + ch) * WK * WC;
#pragma unroll
  for (int ct = 0; ct < 16; ++ct)
    store_cd_f32(pout + (w * 16) * WC + ct * 16, WC, acc[ct]);
}

// ---------------- K11: from_basis2 over full N (B staged by TDM), scale by clamp(out_w) ----------------
__global__ __launch_bounds__(256) void k_frombasis_full(
    const float* __restrict__ evecs, const bf16_t* __restrict__ specT,
    const float* __restrict__ outw, float* __restrict__ out) {
  __shared__ bf16_t aS[128 * 136];  // [n][k]
  __shared__ bf16_t bS[256 * 136];  // [c][k]  (TDM)
  int b = blockIdx.y, nt = blockIdx.x; long n0 = (long)nt * 128;
  int t = threadIdx.x, w = t >> 5;
  if (t < 32) tdm_load_2d_bf16(bS, specT + (long)b * WC * WK, WC, WK, WK, 5, 3);
  for (int i = 0; i < 64; ++i) {
    int e = t + 256 * i; int mm = e >> 7, k = e & 127;
    aS[mm * 136 + k] = f2bf(evecs[((long)b * WN + n0 + mm) * WK + k]);
  }
  if (t < 32) tdm_wait();
  __syncthreads();
  v8f zero = {}; v8f acc[16];
#pragma unroll
  for (int i = 0; i < 16; ++i) acc[i] = zero;
  for (int k0 = 0; k0 < 128; k0 += 32) {
    v16bf af = load_a(&aS[(w * 16) * 136 + k0], 136);
    v16bf bcur = load_b(&bS[k0], 136);
#pragma unroll
    for (int ct = 0; ct < 16; ++ct) {
      v16bf bnxt = bcur;
      if (ct < 15) bnxt = load_b(&bS[((ct + 1) * 16) * 136 + k0], 136);
      acc[ct] = wmma_bf16(af, bcur, acc[ct]);
      bcur = bnxt;
    }
  }
  int lane = t & 31, col = lane & 15, r0 = (lane >> 4) * 8;
  float* o = out + ((long)b * WN + n0) * WC;
#pragma unroll
  for (int ct = 0; ct < 16; ++ct) {
    int c = ct * 16 + col; float ow = fmaxf(outw[c], 1e-8f);
#pragma unroll
    for (int j = 0; j < 8; ++j)
      o[(long)(w * 16 + r0 + j) * WC + c] = acc[ct][j] * ow;
  }
}

// ---------------- host launch ----------------
extern "C" void kernel_launch(void* const* d_in, const int* in_sizes, int n_in,
                              void* d_out, int out_size, void* d_ws, size_t ws_size,
                              hipStream_t stream) {
  const float* x      = (const float*)d_in[0];
  const float* mass   = (const float*)d_in[1];
  const float* evals  = (const float*)d_in[2];
  const float* evecs  = (const float*)d_in[3];
  const int*   far    = (const int*)d_in[4];
  const float* t_in   = (const float*)d_in[5];
  const float* t_out  = (const float*)d_in[6];
  const float* gw     = (const float*)d_in[7];
  const float* gb     = (const float*)d_in[8];
  const float* Wq     = (const float*)d_in[9];
  const float* bq     = (const float*)d_in[10];
  const float* Wk     = (const float*)d_in[11];
  const float* bk     = (const float*)d_in[12];
  const float* Wv     = (const float*)d_in[13];
  const float* bv     = (const float*)d_in[14];
  const float* Wo     = (const float*)d_in[15];
  const float* bo     = (const float*)d_in[16];
  const float* outw   = (const float*)d_in[17];
  float* out = (float*)d_out;

  char* ws = (char*)d_ws;
  float*  part1 = (float*)(ws);                         // 33,554,432 B
  bf16_t* spec1 = (bf16_t*)(ws + 33554432);             //    262,144 B  (transposed [b][c][k])
  float*  xfar  = (float*)(ws + 33816576);              //  4,194,304 B
  float*  stats = (float*)(ws + 38010880);              //      1,024 B
  bf16_t* xn    = (bf16_t*)(ws + 38011904);             //  2,097,152 B
  bf16_t* qb    = (bf16_t*)(ws + 40109056);             //  2,097,152 B
  bf16_t* kb    = (bf16_t*)(ws + 42206208);             //  2,097,152 B
  bf16_t* vb    = (bf16_t*)(ws + 44303360);             //  2,097,152 B
  bf16_t* ao    = (bf16_t*)(ws + 46400512);             //  2,097,152 B
  bf16_t* yfar  = (bf16_t*)(ws + 48497664);             //  2,097,152 B
  float*  part2 = (float*)(ws + 50594816);              //  4,194,304 B
  bf16_t* spec2 = (bf16_t*)(ws + 54789120);             //    262,144 B  (transposed)
  bf16_t* wT    = (bf16_t*)(ws + 55051264);             //    524,288 B  (WqT,WkT,WvT,WoT bf16)

  k_tobasis1<<<dim3(NCH1, WB), 256, 0, stream>>>(x, mass, evecs, part1);
  k_wprep<<<1024, 256, 0, stream>>>(Wq, Wk, Wv, Wo, wT);
  k_reduce<<<(WB * WK * WC) / 256, 256, 0, stream>>>(part1, evals, t_in, spec1, NCH1);
  k_frombasis_far<<<dim3(WM / 128, WB), 256, 0, stream>>>(evecs, far, spec1, xfar);
  k_gnstats<<<WB * 32, 256, 0, stream>>>(xfar, stats);
  k_gnapply<<<(WB * WM * WC) / 256, 256, 0, stream>>>(xfar, stats, gw, gb, xn);
  k_qkv<<<dim3(WM / 128, 3, WB), 256, 0, stream>>>(xn, wT, bq, bk, bv, qb, kb, vb);
  k_attn<<<dim3(WM / 32, WH, WB), 256, 0, stream>>>(qb, kb, vb, ao);
  k_oproj<<<dim3(WM / 128, WB), 256, 0, stream>>>(ao, wT + 3L * 65536, bo, mass, far, yfar);
  k_tobasis2<<<dim3(NCH2, WB), 256, 0, stream>>>(yfar, evecs, far, part2);
  k_reduce<<<(WB * WK * WC) / 256, 256, 0, stream>>>(part2, evals, t_out, spec2, NCH2);
  k_frombasis_full<<<dim3(WN / 128, WB), 256, 0, stream>>>(evecs, spec2, outw, out);
}
